// trajectory2seq_attn_bi_87857851007522
// MI455X (gfx1250) — matrix operand order repaired and checked by reading, not compiled, not run on hardware
//
#include <hip/hip_runtime.h>
#include <math.h>

// ---------------------------------------------------------------------------
// trajectory2seq (bi-LSTM encoder + attention GRU decoder) for MI455X/gfx1250
// All large GEMMs use v_wmma_f32_16x16x32_bf16 (wave32 WMMA), recurrences run
// as persistent workgroups with A-fragment-packed hidden state in LDS.
// ---------------------------------------------------------------------------

typedef __attribute__((ext_vector_type(16))) __bf16 bfrag;  // A or B fragment
typedef __attribute__((ext_vector_type(8)))  float  ffrag;  // C/D fragment

__device__ __forceinline__ ffrag wmma_bf16(bfrag a, bfrag b, ffrag c) {
  return __builtin_amdgcn_wmma_f32_16x16x32_bf16(false, a, false, b, (short)0, c,
                                                 false, false);
}

// Defeat LICM: make a byte/element OFFSET opaque (not the pointer!) so the
// loop-invariant weight-fragment loads are not hoisted out of the time loop
// (hoist => massive VGPR spill), while the base pointer keeps its global
// address space (=> global_load_b128 with SADDR, LOADcnt only — no flat/DScnt).
#define OPAQUE_OFF(o)                                      \
  {                                                        \
    unsigned long long _u = (unsigned long long)(o);       \
    asm volatile("" : "+s"(_u));                           \
    (o) = (size_t)_u;                                      \
  }

__device__ __forceinline__ unsigned short f2bf(float f) {
  union { float f; unsigned u; } x; x.f = f;
  unsigned u = x.u;
  u += 0x7fffu + ((u >> 16) & 1u);          // round-to-nearest-even
  return (unsigned short)(u >> 16);
}
__device__ __forceinline__ float bf2f(unsigned short h) {
  union { float f; unsigned u; } x; x.u = ((unsigned)h) << 16;
  return x.f;
}
__device__ __forceinline__ float sigm(float v) { return 1.0f / (1.0f + __expf(-v)); }

// Position of element (m, kk) inside a packed 16x32 bf16 WMMA *A* tile
// (tile stored as [lane 0..31][elem 0..15], per ISA 7.12.2 layout).
__device__ __forceinline__ int aposf(int m, int kk) {
  return ((m + (((kk & 15) >= 8) ? 16 : 0)) << 4) + ((kk >= 16) ? 8 : 0) + (kk & 7);
}
// Position of element (n, k) inside a packed 32x16 bf16 WMMA *B* tile.
__device__ __forceinline__ int bposf(int n, int k) {
  return ((n + (((k & 15) >= 8) ? 16 : 0)) << 4) + ((k >= 16) ? 8 : 0) + (k & 7);
}

// ---------------------------------------------------------------------------
// Weight packing: src is (N, K) row-major f32; dst is tiles [nt][kt][512] bf16
// holding B(k,n) = src[n][k]  (i.e. the GEMM computes act @ src^T).
// ---------------------------------------------------------------------------
__global__ void k_packB(const float* __restrict__ src, unsigned short* __restrict__ dst,
                        int N, int K) {
  int idx = blockIdx.x * 256 + threadIdx.x;
  if (idx >= N * K) return;
  int n = idx / K, k = idx % K;
  dst[((n >> 4) * (K >> 5) + (k >> 5)) * 512 + bposf(n & 15, k & 31)] = f2bf(src[idx]);
}

// Concat pack: columns [0,Ka) from sA (N,Ka), columns [Ka,Ka+Kb) from sB (N,Kb).
__global__ void k_packB2(const float* __restrict__ sA, int Ka,
                         const float* __restrict__ sB, int Kb,
                         unsigned short* __restrict__ dst, int N) {
  int K = Ka + Kb;
  int idx = blockIdx.x * 256 + threadIdx.x;
  if (idx >= N * K) return;
  int n = idx / K, k = idx % K;
  float v = (k < Ka) ? sA[n * Ka + k] : sB[n * Kb + (k - Ka)];
  dst[((n >> 4) * (K >> 5) + (k >> 5)) * 512 + bposf(n & 15, k & 31)] = f2bf(v);
}

// ---------------------------------------------------------------------------
// LSTM layer 0: persistent workgroup per direction. x is (B,2,T).
// Per step: g = h @ Whh^T (WMMA, K=256) + x_t @ Wih^T (2 MACs) + biases.
// h kept as bf16 A-fragments in ping-pong LDS; c in registers.
// Output y packed as layer-1 A fragments in yPk[t][bt][kt(16)][512] (k = dir*256+j).
// ---------------------------------------------------------------------------
__global__ __launch_bounds__(1024) void k_lstm0(
    const float* __restrict__ x, const float* __restrict__ Wih,
    const float* __restrict__ bih, const float* __restrict__ bhh,
    const unsigned short* __restrict__ WhhPk, unsigned short* __restrict__ yPk,
    float* __restrict__ hn) {
  __shared__ unsigned short hA[2][32768];           // 2 x 64KB ping-pong
  const int dir = blockIdx.x;
  const int tid = threadIdx.x;
  const int wave = tid >> 5, lane = tid & 31;
  const int jt = wave & 15, bq = wave >> 4;
  const int ln = lane & 15, lh = lane >> 4;
  const int j = jt * 16 + ln;                       // hidden index owned by thread

  for (int i = tid; i < 32768; i += 1024) hA[0][i] = 0;
  float c[4][8];
#pragma unroll
  for (int bl = 0; bl < 4; ++bl)
#pragma unroll
    for (int r = 0; r < 8; ++r) c[bl][r] = 0.f;

  float bsum[4], w0[4], w1[4];
#pragma unroll
  for (int g = 0; g < 4; ++g) {
    int n = g * 256 + j;
    bsum[g] = bih[dir * 1024 + n] + bhh[dir * 1024 + n];
    w0[g]   = Wih[(dir * 1024 + n) * 2 + 0];
    w1[g]   = Wih[(dir * 1024 + n) * 2 + 1];
  }
  __syncthreads();

  for (int s = 0; s < 1024; ++s) {
    const int t = dir ? (1023 - s) : s;
    const unsigned short* cur = hA[s & 1];
    unsigned short* nxt = hA[(s + 1) & 1];
    // Opaque offset per step: weight loads stay in-loop (L2-hot) and remain
    // global_load (SADDR form), instead of hoist-then-spill-to-scratch.
    size_t wo = (size_t)dir * (64 * 8 * 512);
    OPAQUE_OFF(wo);
    const unsigned short* WPk = WhhPk + wo;
#pragma unroll
    for (int bl = 0; bl < 4; ++bl) {
      const int bt = bq * 4 + bl;
      ffrag acc[4];
#pragma unroll
      for (int g = 0; g < 4; ++g) acc[g] = (ffrag){0.f,0.f,0.f,0.f,0.f,0.f,0.f,0.f};
#pragma unroll 1
      for (int kt = 0; kt < 8; ++kt) {
        bfrag af = *reinterpret_cast<const bfrag*>(cur + (bt * 8 + kt) * 512 + lane * 16);
#pragma unroll
        for (int g = 0; g < 4; ++g) {
          bfrag bf = *reinterpret_cast<const bfrag*>(
              WPk + ((g * 16 + jt) * 8 + kt) * 512 + lane * 16);
          acc[g] = wmma_bf16(af, bf, acc[g]);
        }
      }
#pragma unroll
      for (int r = 0; r < 8; ++r) {
        int b = bt * 16 + r + 8 * lh;
        float x0 = x[(b * 2 + 0) * 1024 + t];
        float x1 = x[(b * 2 + 1) * 1024 + t];
        float gi = acc[0][r] + bsum[0] + x0 * w0[0] + x1 * w1[0];
        float gf = acc[1][r] + bsum[1] + x0 * w0[1] + x1 * w1[1];
        float gg = acc[2][r] + bsum[2] + x0 * w0[2] + x1 * w1[2];
        float go = acc[3][r] + bsum[3] + x0 * w0[3] + x1 * w1[3];
        float cc = sigm(gf) * c[bl][r] + sigm(gi) * tanhf(gg);
        c[bl][r] = cc;
        float h = sigm(go) * tanhf(cc);
        unsigned short hb = f2bf(h);
        int m = r + 8 * lh;
        nxt[(bt * 8 + (j >> 5)) * 512 + aposf(m, j & 31)] = hb;
        int k2 = dir * 256 + j;  // concat position inside layer-1 input
        yPk[((t * 8 + bt) * 16 + (k2 >> 5)) * 512 + aposf(m, k2 & 31)] = hb;
        if (s == 1023) hn[dir * 32768 + b * 256 + j] = h;
      }
    }
    __syncthreads();
  }
}

// ---------------------------------------------------------------------------
// LSTM layer 1: input projection fused into the recurrence GEMM (K = 512+256).
// A fragments: kt<16 from yPk (global, L2-hot); kt>=16 from LDS hidden state.
// B: packed [Wih1 ; Whh1] (K=768). Output -> y1Pk (enc_y A-fragments) + hn.
// ---------------------------------------------------------------------------
__global__ __launch_bounds__(1024) void k_lstm1(
    const unsigned short* __restrict__ yPk,
    const float* __restrict__ bih, const float* __restrict__ bhh,
    const unsigned short* __restrict__ WPkAll, unsigned short* __restrict__ y1Pk,
    float* __restrict__ hn) {
  __shared__ unsigned short hA[2][32768];
  const int dir = blockIdx.x;
  const int tid = threadIdx.x;
  const int wave = tid >> 5, lane = tid & 31;
  const int jt = wave & 15, bq = wave >> 4;
  const int ln = lane & 15, lh = lane >> 4;
  const int j = jt * 16 + ln;

  for (int i = tid; i < 32768; i += 1024) hA[0][i] = 0;
  float c[4][8];
#pragma unroll
  for (int bl = 0; bl < 4; ++bl)
#pragma unroll
    for (int r = 0; r < 8; ++r) c[bl][r] = 0.f;

  float bsum[4];
#pragma unroll
  for (int g = 0; g < 4; ++g) {
    int n = g * 256 + j;
    bsum[g] = bih[dir * 1024 + n] + bhh[dir * 1024 + n];
  }
  __syncthreads();

  for (int s = 0; s < 1024; ++s) {
    const int t = dir ? (1023 - s) : s;
    const unsigned short* cur = hA[s & 1];
    unsigned short* nxt = hA[(s + 1) & 1];
    size_t wo = (size_t)dir * (64 * 24 * 512);
    OPAQUE_OFF(wo);
    const unsigned short* WPk = WPkAll + wo;
    size_t yo = (size_t)t * (8 * 16 * 512);
    OPAQUE_OFF(yo);
    const unsigned short* yRow = yPk + yo;
#pragma unroll
    for (int bl = 0; bl < 4; ++bl) {
      const int bt = bq * 4 + bl;
      ffrag acc[4];
#pragma unroll
      for (int g = 0; g < 4; ++g) acc[g] = (ffrag){0.f,0.f,0.f,0.f,0.f,0.f,0.f,0.f};
#pragma unroll 1
      for (int kt = 0; kt < 16; ++kt) {          // encoder-layer-0 input part
        bfrag af = *reinterpret_cast<const bfrag*>(yRow + (bt * 16 + kt) * 512 + lane * 16);
#pragma unroll
        for (int g = 0; g < 4; ++g) {
          bfrag bf = *reinterpret_cast<const bfrag*>(
              WPk + ((g * 16 + jt) * 24 + kt) * 512 + lane * 16);
          acc[g] = wmma_bf16(af, bf, acc[g]);
        }
      }
#pragma unroll 1
      for (int kt = 16; kt < 24; ++kt) {         // recurrent part (LDS)
        bfrag af = *reinterpret_cast<const bfrag*>(cur + (bt * 8 + (kt - 16)) * 512 + lane * 16);
#pragma unroll
        for (int g = 0; g < 4; ++g) {
          bfrag bf = *reinterpret_cast<const bfrag*>(
              WPk + ((g * 16 + jt) * 24 + kt) * 512 + lane * 16);
          acc[g] = wmma_bf16(af, bf, acc[g]);
        }
      }
#pragma unroll
      for (int r = 0; r < 8; ++r) {
        int b = bt * 16 + r + 8 * lh;
        float gi = acc[0][r] + bsum[0];
        float gf = acc[1][r] + bsum[1];
        float gg = acc[2][r] + bsum[2];
        float go = acc[3][r] + bsum[3];
        float cc = sigm(gf) * c[bl][r] + sigm(gi) * tanhf(gg);
        c[bl][r] = cc;
        float h = sigm(go) * tanhf(cc);
        unsigned short hb = f2bf(h);
        int m = r + 8 * lh;
        nxt[(bt * 8 + (j >> 5)) * 512 + aposf(m, j & 31)] = hb;
        int k2 = dir * 256 + j;
        y1Pk[((t * 8 + bt) * 16 + (k2 >> 5)) * 512 + aposf(m, k2 & 31)] = hb;
        if (s == 1023) hn[(2 + dir) * 32768 + b * 256 + j] = h;
      }
    }
    __syncthreads();
  }
}

// ---------------------------------------------------------------------------
// fc_hidden: h_out = (h_n.view(B,-1) @ W^T + b).view(L,B,H)  (one-shot, VALU)
// Output flat order equals (L,B,H) row-major == proj flat order (reshape trick).
// ---------------------------------------------------------------------------
__global__ void k_fch(const float* __restrict__ hn, const float* __restrict__ W,
                      const float* __restrict__ bias, float* __restrict__ decHid) {
  int gid = blockIdx.x * 256 + threadIdx.x;   // 65536 outputs
  int row = gid >> 9, col = gid & 511;
  const float* hr = hn + row * 1024;
  const float* wr = W + col * 1024;
  float s = bias[col];
  for (int k = 0; k < 1024; ++k) s += hr[k] * wr[k];
  decHid[gid] = s;
}

// ---------------------------------------------------------------------------
// v_out = enc_y @ h2v_W^T + h2v_b : WMMA GEMM, M = B*T, K = 512, N = 256.
// grid.x = t*8 + bt ; 8 waves, 2 N-tiles each. Output bf16 [b][t][h].
// ---------------------------------------------------------------------------
__global__ __launch_bounds__(256) void k_vout(
    const unsigned short* __restrict__ y1Pk, const unsigned short* __restrict__ h2vPk,
    const float* __restrict__ h2vb, unsigned short* __restrict__ vOut) {
  int blk = blockIdx.x;
  int t = blk >> 3, bt = blk & 7;
  int wave = threadIdx.x >> 5, lane = threadIdx.x & 31;
  int ln = lane & 15, lh = lane >> 4;
#pragma unroll
  for (int nh = 0; nh < 2; ++nh) {
    int nt = wave * 2 + nh;
    ffrag acc = (ffrag){0.f,0.f,0.f,0.f,0.f,0.f,0.f,0.f};
#pragma unroll 1
    for (int kt = 0; kt < 16; ++kt) {
      bfrag af = *reinterpret_cast<const bfrag*>(
          y1Pk + ((t * 8 + bt) * 16 + kt) * 512 + lane * 16);
      bfrag bf = *reinterpret_cast<const bfrag*>(h2vPk + (nt * 16 + kt) * 512 + lane * 16);
      acc = wmma_bf16(af, bf, acc);
    }
    float bias = h2vb[nt * 16 + ln];
#pragma unroll
    for (int r = 0; r < 8; ++r) {
      int b = bt * 16 + r + 8 * lh;
      vOut[(b * 1024 + t) * 256 + nt * 16 + ln] = f2bf(acc[r] + bias);
    }
  }
}

// ---------------------------------------------------------------------------
// One GRU cell over the whole batch (WMMA): gi = in @ Wi^T, gh = h @ Wh^T (N=768).
// Ping-pong packed hidden state (hAcur -> hAnxt), f32 state in hF.
// ---------------------------------------------------------------------------
__device__ __forceinline__ void gru_cell(
    const unsigned short* __restrict__ Ain, const unsigned short* __restrict__ hAcur,
    unsigned short* __restrict__ hAnxt, float* __restrict__ hF,
    const unsigned short* __restrict__ WiPk, const unsigned short* __restrict__ WhPk,
    const float* __restrict__ bi, const float* __restrict__ bhv,
    int lane, int jt, int bq, int lh, int j) {
#pragma unroll
  for (int bl = 0; bl < 4; ++bl) {
    const int bt = bq * 4 + bl;
    ffrag gi[3], gh[3];
#pragma unroll
    for (int g = 0; g < 3; ++g) {
      gi[g] = (ffrag){0.f,0.f,0.f,0.f,0.f,0.f,0.f,0.f};
      gh[g] = (ffrag){0.f,0.f,0.f,0.f,0.f,0.f,0.f,0.f};
    }
#pragma unroll 1
    for (int kt = 0; kt < 8; ++kt) {
      bfrag aI = *reinterpret_cast<const bfrag*>(Ain + (bt * 8 + kt) * 512 + lane * 16);
      bfrag aH = *reinterpret_cast<const bfrag*>(hAcur + (bt * 8 + kt) * 512 + lane * 16);
#pragma unroll
      for (int g = 0; g < 3; ++g) {
        bfrag bI = *reinterpret_cast<const bfrag*>(
            WiPk + ((g * 16 + jt) * 8 + kt) * 512 + lane * 16);
        bfrag bH = *reinterpret_cast<const bfrag*>(
            WhPk + ((g * 16 + jt) * 8 + kt) * 512 + lane * 16);
        gi[g] = wmma_bf16(aI, bI, gi[g]);
        gh[g] = wmma_bf16(aH, bH, gh[g]);
      }
    }
    float br = bi[0 * 256 + j], bz = bi[1 * 256 + j], bn = bi[2 * 256 + j];
    float cr = bhv[0 * 256 + j], cz = bhv[1 * 256 + j], cn = bhv[2 * 256 + j];
#pragma unroll
    for (int r = 0; r < 8; ++r) {
      int b = bt * 16 + r + 8 * lh;
      float rr = sigm(gi[0][r] + br + gh[0][r] + cr);
      float zz = sigm(gi[1][r] + bz + gh[1][r] + cz);
      float nn = tanhf(gi[2][r] + bn + rr * (gh[2][r] + cn));
      float hp = hF[b * 256 + j];
      float hnew = (1.f - zz) * nn + zz * hp;
      hF[b * 256 + j] = hnew;
      hAnxt[(bt * 8 + (j >> 5)) * 512 + aposf(r + 8 * lh, j & 31)] = f2bf(hnew);
    }
  }
}

// ---------------------------------------------------------------------------
// Decoder: single persistent workgroup (1024 threads), 64 sequential steps.
// GRU + attc via WMMA; attention (batched mat-vec) + fc + argmax via VALU.
// ---------------------------------------------------------------------------
__global__ __launch_bounds__(1024) void k_dec(
    const float* __restrict__ emb, const float* __restrict__ gbi,
    const float* __restrict__ gbh, const unsigned short* __restrict__ gWiPk,
    const unsigned short* __restrict__ gWhPk, const unsigned short* __restrict__ attcPkIn,
    const float* __restrict__ attcb, const float* __restrict__ fcW,
    const float* __restrict__ fcb, const unsigned short* __restrict__ vOut,
    float* __restrict__ decHid, unsigned short* inpA, unsigned short* h0A,
    unsigned short* h1A, unsigned short* catA, float* scoresF, float* attF,
    float* attcOut, int* vecIn, float* out) {
  const int tid = threadIdx.x;
  const int wave = tid >> 5, lane = tid & 31;
  const int jt = wave & 15, bq = wave >> 4;
  const int ln = lane & 15, lh = lane >> 4;
  const int j = jt * 16 + ln;

  if (tid < 128) vecIn[tid] = 0;
  for (int idx = tid; idx < 32768; idx += 1024) {
    int b = idx >> 8, jj = idx & 255;
    int p = ((b >> 4) * 8 + (jj >> 5)) * 512 + aposf(b & 15, jj & 31);
    h0A[p] = f2bf(decHid[idx]);
    h1A[p] = f2bf(decHid[32768 + idx]);
  }
  __threadfence(); __syncthreads();

  const float* qF = decHid + 32768;   // top-layer GRU state == q

  for (int step = 0; step < 64; ++step) {
    const int p = step & 1;
    unsigned short* h0cur = h0A + p * 32768;       unsigned short* h0nxt = h0A + (1 - p) * 32768;
    unsigned short* h1cur = h1A + p * 32768;       unsigned short* h1nxt = h1A + (1 - p) * 32768;

    // Opaque offsets per step: all weight-fragment loads stay inside the step
    // loop and keep the global (SADDR) form.
    size_t o0 = 0;               OPAQUE_OFF(o0);
    size_t o1 = 48 * 8 * 512;    OPAQUE_OFF(o1);
    size_t oa = 0;               OPAQUE_OFF(oa);
    const unsigned short* Wi0 = gWiPk + o0;
    const unsigned short* Wh0 = gWhPk + o0;
    const unsigned short* Wi1 = gWiPk + o1;
    const unsigned short* Wh1 = gWhPk + o1;
    const unsigned short* attcPk = attcPkIn + oa;

    // 1) embedding lookup -> packed A fragments
    for (int idx = tid; idx < 32768; idx += 1024) {
      int b = idx >> 8, k = idx & 255;
      float v = emb[vecIn[b] * 256 + k];
      inpA[((b >> 4) * 8 + (k >> 5)) * 512 + aposf(b & 15, k & 31)] = f2bf(v);
    }
    __threadfence(); __syncthreads();

    // 2) two GRU cells (WMMA)
    gru_cell(inpA, h0cur, h0nxt, decHid, Wi0, Wh0, gbi, gbh,
             lane, jt, bq, lh, j);
    __threadfence(); __syncthreads();
    gru_cell(h0nxt, h1cur, h1nxt, decHid + 32768,
             Wi1, Wh1, gbi + 768, gbh + 768,
             lane, jt, bq, lh, j);
    __threadfence(); __syncthreads();

    // 3) attention scores: s[b,t] = v_out[b,t,:] . q[b,:]
    for (int idx = tid; idx < 131072; idx += 1024) {
      int b = idx >> 10, t = idx & 1023;
      const unsigned short* vp = vOut + (b * 1024 + t) * 256;
      const float* qp = qF + b * 256;
      float s = 0.f;
      for (int h = 0; h < 256; ++h) s += qp[h] * bf2f(vp[h]);
      scoresF[idx] = s;
    }
    __threadfence(); __syncthreads();

    // 4) softmax over t (per b; each wave owns 4 rows, lane-parallel + shfl)
    for (int bb = 0; bb < 4; ++bb) {
      int b = wave * 4 + bb;
      float* sp = scoresF + b * 1024;
      float mx = -1e30f;
      for (int t = lane; t < 1024; t += 32) mx = fmaxf(mx, sp[t]);
      for (int o = 16; o > 0; o >>= 1) mx = fmaxf(mx, __shfl_xor(mx, o, 32));
      float sm = 0.f;
      for (int t = lane; t < 1024; t += 32) sm += __expf(sp[t] - mx);
      for (int o = 16; o > 0; o >>= 1) sm += __shfl_xor(sm, o, 32);
      float inv = 1.f / sm;
      for (int t = lane; t < 1024; t += 32) sp[t] = __expf(sp[t] - mx) * inv;
    }
    __threadfence(); __syncthreads();

    // 5) att[b,h] = sum_t aw[b,t] * v_out[b,t,h]
    for (int idx = tid; idx < 32768; idx += 1024) {
      int b = idx >> 8, h = idx & 255;
      const float* ap = scoresF + b * 1024;
      const unsigned short* vp = vOut + (size_t)b * 1024 * 256 + h;
      float s = 0.f;
      for (int t = 0; t < 1024; ++t) s += ap[t] * bf2f(vp[t * 256]);
      attF[idx] = s;
    }
    __threadfence(); __syncthreads();

    // 6) cat = [q, att] -> packed A fragments (K = 512)
    for (int idx = tid; idx < 65536; idx += 1024) {
      int b = idx >> 9, k = idx & 511;
      float v = (k < 256) ? qF[b * 256 + k] : attF[b * 256 + (k - 256)];
      catA[((b >> 4) * 16 + (k >> 5)) * 512 + aposf(b & 15, k & 31)] = f2bf(v);
    }
    __threadfence(); __syncthreads();

    // 7) attc GEMM: (128x512) @ (512x256) via WMMA
    {
      int nt = wave & 15, bq2 = wave >> 4;
#pragma unroll
      for (int bl = 0; bl < 4; ++bl) {
        int bt = bq2 * 4 + bl;
        ffrag acc = (ffrag){0.f,0.f,0.f,0.f,0.f,0.f,0.f,0.f};
#pragma unroll 1
        for (int kt = 0; kt < 16; ++kt) {
          bfrag af = *reinterpret_cast<const bfrag*>(catA + (bt * 16 + kt) * 512 + lane * 16);
          bfrag bf = *reinterpret_cast<const bfrag*>(attcPk + (nt * 16 + kt) * 512 + lane * 16);
          acc = wmma_bf16(af, bf, acc);
        }
        float bias = attcb[nt * 16 + ln];
#pragma unroll
        for (int r = 0; r < 8; ++r) {
          int b = bt * 16 + r + 8 * lh;
          attcOut[b * 256 + nt * 16 + ln] = acc[r] + bias;
        }
      }
    }
    __threadfence(); __syncthreads();

    // 8) logits = attcOut @ fc_W^T + fc_b, written straight into vec_out
    for (int idx = tid; idx < 3840; idx += 1024) {
      int b = idx / 30, v = idx % 30;
      const float* ap = attcOut + b * 256;
      const float* wp = fcW + v * 256;
      float s = fcb[v];
      for (int h = 0; h < 256; ++h) s += ap[h] * wp[h];
      out[b * 1920 + step * 30 + v] = s;
    }
    __threadfence(); __syncthreads();

    // 9) argmax feedback (first max wins, like jnp.argmax)
    if (tid < 128) {
      const float* lp = out + tid * 1920 + step * 30;
      float best = lp[0]; int bi_ = 0;
      for (int v = 1; v < 30; ++v) { float vv = lp[v]; if (vv > best) { best = vv; bi_ = v; } }
      vecIn[tid] = bi_;
    }
    __threadfence(); __syncthreads();
  }

  // hidden_T (L,B,H) and last-step attention weights (B,1,T)
  for (int idx = tid; idx < 65536; idx += 1024) out[245760 + idx] = decHid[idx];
  for (int idx = tid; idx < 131072; idx += 1024) out[245760 + 65536 + idx] = scoresF[idx];
}

// ---------------------------------------------------------------------------
extern "C" void kernel_launch(void* const* d_in, const int* in_sizes, int n_in,
                              void* d_out, int out_size, void* d_ws, size_t ws_size,
                              hipStream_t stream) {
  const float* x     = (const float*)d_in[0];
  const float* l0Wih = (const float*)d_in[1];
  const float* l0Whh = (const float*)d_in[2];
  const float* l0bih = (const float*)d_in[3];
  const float* l0bhh = (const float*)d_in[4];
  const float* l1Wih = (const float*)d_in[5];
  const float* l1Whh = (const float*)d_in[6];
  const float* l1bih = (const float*)d_in[7];
  const float* l1bhh = (const float*)d_in[8];
  const float* fchW  = (const float*)d_in[9];
  const float* fchB  = (const float*)d_in[10];
  const float* emb   = (const float*)d_in[11];
  const float* gWi   = (const float*)d_in[12];
  const float* gWh   = (const float*)d_in[13];
  const float* gbi   = (const float*)d_in[14];
  const float* gbh   = (const float*)d_in[15];
  const float* attcW = (const float*)d_in[16];
  const float* attcb = (const float*)d_in[17];
  const float* h2vW  = (const float*)d_in[18];
  const float* h2vb  = (const float*)d_in[19];
  const float* fcW   = (const float*)d_in[20];
  const float* fcb   = (const float*)d_in[21];
  float* out = (float*)d_out;
  (void)in_sizes; (void)n_in; (void)out_size; (void)ws_size;

  char* ws = (char*)d_ws;
  size_t off = 0;
  auto alloc = [&](size_t bytes) -> void* {
    void* p = ws + off;
    off += (bytes + 1023) & ~(size_t)1023;
    return p;
  };
  unsigned short* yPk    = (unsigned short*)alloc(134217728);        // (T,8,16,512) bf16
  unsigned short* y1Pk   = (unsigned short*)alloc(134217728);
  unsigned short* vOut   = (unsigned short*)alloc(67108864);         // (B,T,H) bf16
  unsigned short* WhhPk0 = (unsigned short*)alloc((size_t)2*64*8*512*2);
  unsigned short* WcPk1  = (unsigned short*)alloc((size_t)2*64*24*512*2);
  unsigned short* gWiPk  = (unsigned short*)alloc((size_t)2*48*8*512*2);
  unsigned short* gWhPk  = (unsigned short*)alloc((size_t)2*48*8*512*2);
  unsigned short* attcPk = (unsigned short*)alloc((size_t)16*16*512*2);
  unsigned short* h2vPk  = (unsigned short*)alloc((size_t)16*16*512*2);
  float* hn      = (float*)alloc((size_t)4*32768*4);
  float* decHid  = (float*)alloc((size_t)2*32768*4);
  unsigned short* inpA = (unsigned short*)alloc((size_t)32768*2);
  unsigned short* h0A  = (unsigned short*)alloc((size_t)2*32768*2);
  unsigned short* h1A  = (unsigned short*)alloc((size_t)2*32768*2);
  unsigned short* catA = (unsigned short*)alloc((size_t)65536*2);
  float* scoresF = (float*)alloc((size_t)131072*4);
  float* attF    = (float*)alloc((size_t)32768*4);
  float* attcOut = (float*)alloc((size_t)32768*4);
  int*   vecIn   = (int*)alloc(512);

  // ---- weight packing (B-fragment layout, bf16) ----
  k_packB <<<1024, 256, 0, stream>>>(l0Whh,          WhhPk0,              1024, 256);
  k_packB <<<1024, 256, 0, stream>>>(l0Whh + 262144, WhhPk0 + 64*8*512,   1024, 256);
  k_packB2<<<3072, 256, 0, stream>>>(l1Wih,          512, l1Whh,          256, WcPk1,              1024);
  k_packB2<<<3072, 256, 0, stream>>>(l1Wih + 524288, 512, l1Whh + 262144, 256, WcPk1 + 64*24*512,  1024);
  k_packB <<<768, 256, 0, stream>>>(gWi,           gWiPk,            768, 256);
  k_packB <<<768, 256, 0, stream>>>(gWi + 196608,  gWiPk + 48*8*512, 768, 256);
  k_packB <<<768, 256, 0, stream>>>(gWh,           gWhPk,            768, 256);
  k_packB <<<768, 256, 0, stream>>>(gWh + 196608,  gWhPk + 48*8*512, 768, 256);
  k_packB <<<512, 256, 0, stream>>>(attcW, attcPk, 256, 512);
  k_packB <<<512, 256, 0, stream>>>(h2vW,  h2vPk,  256, 512);

  // ---- encoder ----
  k_lstm0<<<2, 1024, 0, stream>>>(x, l0Wih, l0bih, l0bhh, WhhPk0, yPk, hn);
  k_lstm1<<<2, 1024, 0, stream>>>(yPk, l1bih, l1bhh, WcPk1, y1Pk, hn);
  k_fch  <<<256, 256, 0, stream>>>(hn, fchW, fchB, decHid);
  k_vout <<<8192, 256, 0, stream>>>(y1Pk, h2vPk, h2vb, vOut);

  // ---- decoder ----
  k_dec<<<1, 1024, 0, stream>>>(emb, gbi, gbh, gWiPk, gWhPk, attcPk, attcb,
                                fcW, fcb, vOut, decHid, inpA, h0A, h1A, catA,
                                scoresF, attF, attcOut, vecIn, out);
}
